// SchNorbInteraction_27547920237285
// MI455X (gfx1250) — compile-verified, matrix operand order checked by reading, not run on hardware
//
#include <hip/hip_runtime.h>

// ---------------------------------------------------------------------------
// Types for CDNA5 WMMA (wave32)
// ---------------------------------------------------------------------------
typedef __bf16 bf16;
typedef __bf16 v16bf __attribute__((ext_vector_type(16)));
typedef __bf16 v8bf  __attribute__((ext_vector_type(8)));
typedef float  v8f   __attribute__((ext_vector_type(8)));

#define Bc   4
#define Ac   256
#define Nn   64
#define NFc  128
#define NCBc 128
#define NSBc 50

// shifted softplus: softplus(x) - ln 2  (numerically stable)
__device__ __forceinline__ float sspf(float x) {
  float t = __logf(1.0f + __expf(-fabsf(x)));
  return fmaxf(x, 0.0f) + t - 0.6931471805599453f;
}

// ---------------------------------------------------------------------------
// Async global -> LDS copy (CDNA5 GLOBAL_LOAD_ASYNC_TO_LDS_B128, ASYNCcnt).
// Writes LDS directly, no VGPR round-trip. bytes must be a multiple of
// 256 threads * 16B. LDS address = low 32 bits of the generic shared pointer
// (workgroup-relative LDS offset). Caller must __syncthreads() afterwards.
// ---------------------------------------------------------------------------
__device__ __forceinline__ void async_copy_w(bf16* __restrict__ sW,
                                             const bf16* __restrict__ gsrc,
                                             int bytes, int tid) {
  const unsigned lds_base = (unsigned)(size_t)sW;
  const unsigned long long gbase = (unsigned long long)(size_t)gsrc;
  for (int i = tid * 16; i < bytes; i += 256 * 16) {
    unsigned voff  = (unsigned)i;
    unsigned laddr = lds_base + (unsigned)i;
    asm volatile("global_load_async_to_lds_b128 %0, %1, %2"
                 :
                 : "v"(laddr), "v"(voff), "s"(gbase)
                 : "memory");
  }
  asm volatile("s_wait_asynccnt 0" ::: "memory");
}

// ---------------------------------------------------------------------------
// WMMA operand fragment load from LDS.
// 16-bit A/B layout (ISA 7.12.2): lane half h = lane>>4, row = base + (lane&15),
// element e -> K = kk + (e<8 ? 8h+e : 8+8h+e)  => two contiguous 16B runs.
// ---------------------------------------------------------------------------
__device__ __forceinline__ v16bf load_frag(const bf16* __restrict__ base,
                                           int stride, int row16, int kk, int lane) {
  const int r = row16 + (lane & 15);
  const int h = lane >> 4;
  const bf16* p = base + r * stride + kk + 8 * h;
  v8bf lo = *(const v8bf*)p;
  v8bf hi = *(const v8bf*)(p + 16);
  v16bf f;
#pragma unroll
  for (int i = 0; i < 8; ++i) { f[i] = lo[i]; f[i + 8] = hi[i]; }
  return f;
}

// ---------------------------------------------------------------------------
// One dense layer:  out[64 x 128] = epi( in[64 x K] @ W[K x 128] + bias )
// Weights are pre-transposed bf16 [128 cols][K], streamed async through a
// 16KB LDS slab in two 64-column halves. 8 waves x 2 tiles/half => 16 tiles.
// Epi(mt, col, h, acc) handles the 8 f32 results (rows m = mt + r + 8h).
// ---------------------------------------------------------------------------
template <typename Epi>
__device__ __forceinline__ void dense_layer(const bf16* __restrict__ inbuf, int K,
                                            const bf16* __restrict__ gWT,
                                            const float* __restrict__ bias,
                                            bf16* __restrict__ sW,
                                            int tid, int wave, int lane, Epi epi) {
#pragma unroll
  for (int half = 0; half < 2; ++half) {
    __syncthreads();  // previous users of sW are done
    async_copy_w(sW, gWT + (size_t)half * 64 * K, 64 * K * 2, tid);
    __syncthreads();
    const int colbase = half << 6;
#pragma unroll
    for (int t = 0; t < 2; ++t) {
      const int tile = wave * 2 + t;
      const int mt  = (tile >> 2) << 4;   // row tile base (0..48)
      const int ntl = (tile & 3) << 4;    // local col tile base (0..48)
      const int col = colbase + ntl + (lane & 15);
      const float bv = bias ? bias[col] : 0.0f;
      v8f c;
#pragma unroll
      for (int i = 0; i < 8; ++i) c[i] = bv;
#pragma unroll
      for (int kk = 0; kk < K; kk += 32) {
        v16bf a = load_frag(inbuf, K, mt, kk, lane);
        v16bf b = load_frag(sW,  K, ntl, kk, lane);
        c = __builtin_amdgcn_wmma_f32_16x16x32_bf16(false, a, false, b,
                                                    (short)0, c, false, false);
      }
      epi(mt, col, lane >> 4, c);
    }
  }
  __syncthreads();  // outputs visible; sW free
}

// ---------------------------------------------------------------------------
// Weight convert + transpose:  dst[n*Kpad + k] = (k<K) ? bf16(src[k*Ncols+n]) : 0
// ---------------------------------------------------------------------------
__global__ void convw_kernel(const float* __restrict__ src, bf16* __restrict__ dst,
                             int K, int Ncols, int Kpad) {
  int i = blockIdx.x * blockDim.x + threadIdx.x;
  if (i >= Ncols * Kpad) return;
  int n = i / Kpad;
  int k = i - n * Kpad;
  dst[i] = (bf16)((k < K) ? src[k * Ncols + n] : 0.0f);
}

// ---------------------------------------------------------------------------
// y = xi @ in2f_W   (1024 x 128, K=128), WMMA, 16 blocks x 64 rows
// ---------------------------------------------------------------------------
__global__ __launch_bounds__(256) void y_kernel(const float* __restrict__ xi,
                                                const bf16* __restrict__ wT_in2f,
                                                float* __restrict__ y) {
  __shared__ __align__(16) char smem[32768];
  bf16* sW = (bf16*)smem;
  bf16* sA = (bf16*)(smem + 16384);
  const int tid = threadIdx.x, wave = tid >> 5, lane = tid & 31;
  const size_t rowbase = (size_t)blockIdx.x * 64;
  for (int i = tid; i < 64 * 128; i += 256) sA[i] = (bf16)xi[rowbase * 128 + i];
  dense_layer(sA, 128, wT_in2f, (const float*)nullptr, sW, tid, wave, lane,
      [&](int mt, int col, int h, const v8f& c) {
#pragma unroll
        for (int r = 0; r < 8; ++r) {
          int m = mt + r + 8 * h;
          y[(rowbase + m) * 128 + col] = c[r];
        }
      });
}

// ---------------------------------------------------------------------------
// Pass 1: one workgroup per atom (b,a). Fuses the whole per-pair pipeline:
//   Wf = ssp(f@W1+b1)@W2+b2, cutoff, yj gather, v = ssp((yj*Wf)@f2out+b),
//   vi = mlp2(sum v), vij = mlp2_pair(v) -> ws, Vik = sum vik*cos -> ws
// ---------------------------------------------------------------------------
__global__ __launch_bounds__(256) void pass1_kernel(
    const float* __restrict__ r_ij, const float* __restrict__ cos_ij,
    const float* __restrict__ f_ij, const float* __restrict__ nmask,
    const int* __restrict__ nbrs,
    const float* __restrict__ filt_b1, const float* __restrict__ filt_b2,
    const float* __restrict__ f2out_b,
    const float* __restrict__ atom_W1, const float* __restrict__ atom_b1,
    const float* __restrict__ atom_W2, const float* __restrict__ atom_b2,
    const float* __restrict__ pair_b1, const float* __restrict__ pair_b2,
    const float* __restrict__ env_b1, const float* __restrict__ env_b2,
    const bf16* __restrict__ wT_filt1, const bf16* __restrict__ wT_filt2,
    const bf16* __restrict__ wT_f2out, const bf16* __restrict__ wT_pair1,
    const bf16* __restrict__ wT_pair2, const bf16* __restrict__ wT_env1,
    const bf16* __restrict__ wT_env2,
    const float* __restrict__ y, float* __restrict__ vij,
    float* __restrict__ VikW, float* __restrict__ vi_out) {
  __shared__ __align__(16) char smem[53248];
  bf16*  sW    = (bf16*)smem;            // 16 KB weight slab (async-streamed)
  bf16*  sA    = (bf16*)(smem + 16384);  // 16 KB activation A
  bf16*  sV    = (bf16*)(smem + 32768);  // 16 KB activation B (holds v)
  float* sC    = (float*)(smem + 49152); // cutoff[64]
  float* sMask = (float*)(smem + 49408); // mask[64]
  int*   sNbr  = (int*)  (smem + 49664); // neighbors[64]
  float* sCos  = (float*)(smem + 49920); // cos[64][3]
  float* sVsum = (float*)(smem + 50688); // [128]
  float* sTmp  = (float*)(smem + 51200); // [128]
  float* sVik  = (float*)(smem + 51712); // [128][3]

  const int tid = threadIdx.x, wave = tid >> 5, lane = tid & 31;
  const int atom = blockIdx.x;                 // b*A + a
  const size_t pairbase = (size_t)atom * Nn;
  const int ybase = (atom >> 8) << 8;          // b*A

  if (tid < 64) {
    float r = r_ij[pairbase + tid];
    float c = 0.5f * (__cosf(r * 0.6283185307179586f) + 1.0f);  // pi/5
    sC[tid] = (r < 5.0f) ? c : 0.0f;
    sMask[tid] = nmask[pairbase + tid];
    sNbr[tid]  = nbrs[pairbase + tid];
    sCos[tid * 3 + 0] = cos_ij[(pairbase + tid) * 3 + 0];
    sCos[tid * 3 + 1] = cos_ij[(pairbase + tid) * 3 + 1];
    sCos[tid * 3 + 2] = cos_ij[(pairbase + tid) * 3 + 2];
  }
  for (int i = tid; i < 384; i += 256) sVik[i] = 0.0f;
  // f_ij -> sA [64 rows][K=64], zero-padded past NSB=50
  for (int i = tid; i < 64 * 64; i += 256) {
    int row = i >> 6, k = i & 63;
    sA[i] = (bf16)((k < NSBc) ? f_ij[(pairbase + row) * NSBc + k] : 0.0f);
  }

  // GEMM1: ssp(f @ filt_W1 + b1) -> sV
  dense_layer(sA, 64, wT_filt1, filt_b1, sW, tid, wave, lane,
      [&](int mt, int col, int h, const v8f& c) {
#pragma unroll
        for (int r = 0; r < 8; ++r) {
          int m = mt + r + 8 * h;
          sV[m * 128 + col] = (bf16)sspf(c[r]);
        }
      });

  // GEMM2: Wf = sV @ filt_W2 + b2; h = yj * Wf * C(r) -> sA  (yj gather)
  dense_layer(sV, 128, wT_filt2, filt_b2, sW, tid, wave, lane,
      [&](int mt, int col, int h, const v8f& c) {
#pragma unroll
        for (int r = 0; r < 8; ++r) {
          int m = mt + r + 8 * h;
          float yv = y[(size_t)(ybase + sNbr[m]) * 128 + col];
          sA[m * 128 + col] = (bf16)(c[r] * sC[m] * yv);
        }
      });

  // GEMM3: v = ssp(sA @ f2out_W + b) -> sV
  dense_layer(sA, 128, wT_f2out, f2out_b, sW, tid, wave, lane,
      [&](int mt, int col, int h, const v8f& c) {
#pragma unroll
        for (int r = 0; r < 8; ++r) {
          int m = mt + r + 8 * h;
          sV[m * 128 + col] = (bf16)sspf(c[r]);
        }
      });

  // vi = mlp2(sum_m v*mask)  (tiny: VALU)
  if (tid < 128) {
    float s = 0.0f;
    for (int m = 0; m < 64; ++m) s += (float)sV[m * 128 + tid] * sMask[m];
    sVsum[tid] = s;
  }
  __syncthreads();
  if (tid < 128) {
    float s = atom_b1[tid];
    for (int k = 0; k < 128; ++k) s += sVsum[k] * atom_W1[k * 128 + tid];
    sTmp[tid] = sspf(s);
  }
  __syncthreads();
  if (tid < 128) {
    float s = atom_b2[tid];
    for (int k = 0; k < 128; ++k) s += sTmp[k] * atom_W2[k * 128 + tid];
    vi_out[(size_t)atom * 128 + tid] = s;
  }

  // pair branch: vij = ssp(v@pW1+b)@pW2+b -> global ws
  dense_layer(sV, 128, wT_pair1, pair_b1, sW, tid, wave, lane,
      [&](int mt, int col, int h, const v8f& c) {
#pragma unroll
        for (int r = 0; r < 8; ++r) {
          int m = mt + r + 8 * h;
          sA[m * 128 + col] = (bf16)sspf(c[r]);
        }
      });
  dense_layer(sA, 128, wT_pair2, pair_b2, sW, tid, wave, lane,
      [&](int mt, int col, int h, const v8f& c) {
#pragma unroll
        for (int r = 0; r < 8; ++r) {
          int m = mt + r + 8 * h;
          vij[(pairbase + m) * 128 + col] = c[r];
        }
      });

  // env branch: vik = ssp(v@eW1+b)@eW2+b; Vik[c][d] = sum_m vik*cos*mask
  dense_layer(sV, 128, wT_env1, env_b1, sW, tid, wave, lane,
      [&](int mt, int col, int h, const v8f& c) {
#pragma unroll
        for (int r = 0; r < 8; ++r) {
          int m = mt + r + 8 * h;
          sA[m * 128 + col] = (bf16)sspf(c[r]);
        }
      });
  dense_layer(sA, 128, wT_env2, env_b2, sW, tid, wave, lane,
      [&](int mt, int col, int h, const v8f& c) {
        float p0 = 0.0f, p1 = 0.0f, p2 = 0.0f;
#pragma unroll
        for (int r = 0; r < 8; ++r) {
          int m = mt + r + 8 * h;
          float vv = c[r] * sMask[m];
          p0 += vv * sCos[m * 3 + 0];
          p1 += vv * sCos[m * 3 + 1];
          p2 += vv * sCos[m * 3 + 2];
        }
        atomicAdd(&sVik[col * 3 + 0], p0);
        atomicAdd(&sVik[col * 3 + 1], p1);
        atomicAdd(&sVik[col * 3 + 2], p2);
      });

  for (int i = tid; i < 384; i += 256) VikW[(size_t)atom * 384 + i] = sVik[i];
}

// ---------------------------------------------------------------------------
// Pass 2 (bandwidth limiter): V = vij*cos + Vik_i + Vik_j(gather)  (100 MB out)
// ---------------------------------------------------------------------------
__global__ __launch_bounds__(256) void pass2_kernel(const float* __restrict__ vij,
                                                    const float* __restrict__ Vik,
                                                    const float* __restrict__ cos_ij,
                                                    const int* __restrict__ nbrs,
                                                    float* __restrict__ V) {
  size_t idx = (size_t)blockIdx.x * 256 + threadIdx.x;  // over B*A*N*NCB
  int c = (int)(idx & 127);
  size_t pair = idx >> 7;
  int atom = (int)(pair >> 6);
  int aj = ((atom >> 8) << 8) + nbrs[pair];
  float vv = vij[idx];
  const float* cp = cos_ij + pair * 3;
  const float* pi = Vik + (size_t)atom * 384 + c * 3;
  const float* pj = Vik + (size_t)aj * 384 + c * 3;
  size_t o = idx * 3;
  V[o + 0] = vv * cp[0] + pi[0] + pj[0];
  V[o + 1] = vv * cp[1] + pi[1] + pj[1];
  V[o + 2] = vv * cp[2] + pi[2] + pj[2];
}

// ---------------------------------------------------------------------------
extern "C" void kernel_launch(void* const* d_in, const int* in_sizes, int n_in,
                              void* d_out, int out_size, void* d_ws, size_t ws_size,
                              hipStream_t stream) {
  const float* xi      = (const float*)d_in[0];
  const float* r_ij    = (const float*)d_in[1];
  const float* cos_ij  = (const float*)d_in[2];
  const float* f_ij    = (const float*)d_in[3];
  const float* nmask   = (const float*)d_in[4];
  const float* filt_W1 = (const float*)d_in[5];
  const float* filt_b1 = (const float*)d_in[6];
  const float* filt_W2 = (const float*)d_in[7];
  const float* filt_b2 = (const float*)d_in[8];
  const float* in2f_W  = (const float*)d_in[9];
  const float* f2out_W = (const float*)d_in[10];
  const float* f2out_b = (const float*)d_in[11];
  const float* atom_W1 = (const float*)d_in[12];
  const float* atom_b1 = (const float*)d_in[13];
  const float* atom_W2 = (const float*)d_in[14];
  const float* atom_b2 = (const float*)d_in[15];
  const float* pair_W1 = (const float*)d_in[16];
  const float* pair_b1 = (const float*)d_in[17];
  const float* pair_W2 = (const float*)d_in[18];
  const float* pair_b2 = (const float*)d_in[19];
  const float* env_W1  = (const float*)d_in[20];
  const float* env_b1  = (const float*)d_in[21];
  const float* env_W2  = (const float*)d_in[22];
  const float* env_b2  = (const float*)d_in[23];
  const int*   nbrs    = (const int*)d_in[24];

  char* ws = (char*)d_ws;
  bf16* wT_filt1 = (bf16*)(ws + 0);                 // 128*64  bf16 = 16 KB
  bf16* wT_filt2 = (bf16*)(ws + 16384);             // 128*128 bf16 = 32 KB each
  bf16* wT_in2f  = (bf16*)(ws + 16384 + 32768 * 1);
  bf16* wT_f2out = (bf16*)(ws + 16384 + 32768 * 2);
  bf16* wT_pair1 = (bf16*)(ws + 16384 + 32768 * 3);
  bf16* wT_pair2 = (bf16*)(ws + 16384 + 32768 * 4);
  bf16* wT_env1  = (bf16*)(ws + 16384 + 32768 * 5);
  bf16* wT_env2  = (bf16*)(ws + 16384 + 32768 * 6);
  float* yW   = (float*)(ws + 245760);              // 1024*128 f32
  float* vijW = (float*)(ws + 245760 + 524288);     // 65536*128 f32
  float* VikB = (float*)(ws + 245760 + 524288 + 33554432);  // 1024*384 f32

  float* vi_out = (float*)d_out;
  float* V_out  = vi_out + (size_t)Bc * Ac * NCBc;  // 131072

  convw_kernel<<<32, 256, 0, stream>>>(filt_W1, wT_filt1, 50, 128, 64);
  convw_kernel<<<64, 256, 0, stream>>>(filt_W2, wT_filt2, 128, 128, 128);
  convw_kernel<<<64, 256, 0, stream>>>(in2f_W,  wT_in2f,  128, 128, 128);
  convw_kernel<<<64, 256, 0, stream>>>(f2out_W, wT_f2out, 128, 128, 128);
  convw_kernel<<<64, 256, 0, stream>>>(pair_W1, wT_pair1, 128, 128, 128);
  convw_kernel<<<64, 256, 0, stream>>>(pair_W2, wT_pair2, 128, 128, 128);
  convw_kernel<<<64, 256, 0, stream>>>(env_W1,  wT_env1,  128, 128, 128);
  convw_kernel<<<64, 256, 0, stream>>>(env_W2,  wT_env2,  128, 128, 128);

  y_kernel<<<16, 256, 0, stream>>>(xi, wT_in2f, yW);

  pass1_kernel<<<Bc * Ac, 256, 0, stream>>>(
      r_ij, cos_ij, f_ij, nmask, nbrs,
      filt_b1, filt_b2, f2out_b,
      atom_W1, atom_b1, atom_W2, atom_b2,
      pair_b1, pair_b2, env_b1, env_b2,
      wT_filt1, wT_filt2, wT_f2out, wT_pair1, wT_pair2, wT_env1, wT_env2,
      yW, vijW, VikB, vi_out);

  pass2_kernel<<<(Bc * Ac * Nn * NCBc) / 256, 256, 0, stream>>>(
      vijW, VikB, cos_ij, nbrs, V_out);
}
